// Spatial_Aware_Layer_68822555951426
// MI455X (gfx1250) — matrix operand order repaired and checked by reading, not compiled, not run on hardware
//
#include <hip/hip_runtime.h>

// Problem constants (B=8, C=256, H=W=64, 3x3 deformable conv, DCNv2)
#define CIN   256
#define COUT  256
#define HW    4096
#define KTAP  9
#define KDIM  2304   // CIN * 9
#define OMCH  27     // 18 offset + 9 mask channels
#define KSTEPS 72    // KDIM / 32

typedef __attribute__((ext_vector_type(16))) __bf16 v16bf;
typedef __attribute__((ext_vector_type(8)))  float  v8f;

union FragBF { v16bf v; uint4 q[2]; };

__device__ __forceinline__ unsigned short f2bf(float f) {
    unsigned int u = __float_as_uint(f);
    u += 0x7FFFu + ((u >> 16) & 1u);      // round-to-nearest-even
    return (unsigned short)(u >> 16);
}

// ws layout (bytes):
//   [0,            3538944)  om     : 8*27*4096 f32
//   [3538944,      3686400)  wom_bf : 32*2304 bf16 (27 padded to 32)
//   [3686400,      4866048)  wdc_bf : 256*2304 bf16
#define OM_OFF     0
#define WOMBF_OFF  3538944
#define WDCBF_OFF  3686400
#define WOM_ELEMS  (32 * KDIM)                 // 73728
#define WDC_ELEMS  (COUT * KDIM)               // 589824

// ---------------------------------------------------------------------------
// Kernel 0: one-shot f32 -> bf16 weight conversion (re-run every call,
// deterministic). w_om padded 27 -> 32 output rows.
// ---------------------------------------------------------------------------
__global__ __launch_bounds__(256) void dcn_convert_weights(
    const float* __restrict__ w_om, const float* __restrict__ w_dc,
    unsigned short* __restrict__ wom_bf, unsigned short* __restrict__ wdc_bf)
{
    const int idx = blockIdx.x * 256 + threadIdx.x;
    if (idx < WOM_ELEMS) {
        wom_bf[idx] = f2bf(idx < OMCH * KDIM ? w_om[idx] : 0.f);
    }
    const int j = idx - WOM_ELEMS;
    if (j >= 0 && j < WDC_ELEMS) {
        wdc_bf[j] = f2bf(w_dc[j]);
    }
}

// ---------------------------------------------------------------------------
// Kernel 1: offset/mask 3x3 conv as WMMA GEMM.
// M = 64 pixels per workgroup, N = 32 (27 used), K = 2304 in chunks of 32.
// Grid: (64 pixel-groups, 8 batches), 128 threads = 4 waves.
// ---------------------------------------------------------------------------
__global__ __launch_bounds__(128) void dcn_offset_conv_wmma(
    const float* __restrict__ x, const unsigned short* __restrict__ wom_bf,
    const float* __restrict__ b_om, float* __restrict__ om)
{
    __shared__ __align__(16) unsigned short sA[64][32];  // pixels x kk (bf16)
    __shared__ __align__(16) unsigned short sB[32][32];  // n x kk (bf16)

    const int b = blockIdx.y;
    const int pix_base = blockIdx.x * 64;
    const int tid  = threadIdx.x;
    const int wv   = tid >> 5;
    const int lane = tid & 31;

    v8f acc0 = {};
    v8f acc1 = {};

    const int mrow = lane & 15;
    const int kbA  = (lane < 16) ? 0 : 16;
    const int kbB  = (lane < 16) ? 0 : 16;

    for (int it = 0; it < KSTEPS; ++it) {
        const int kk0 = it * 32;
        // Stage A: shifted-input im2col tile, 64 pixels x 32 kk
        for (int e = tid; e < 2048; e += 128) {
            const int m = e >> 5, i = e & 31;
            const int kk = kk0 + i;
            const int c = kk / 9, k = kk - 9 * c;
            const int p = pix_base + m;
            const int hy = (p >> 6) - 1 + (k / 3);
            const int wx = (p & 63) - 1 + (k - 3 * (k / 3));
            float v = 0.f;
            if (hy >= 0 && hy < 64 && wx >= 0 && wx < 64)
                v = x[((b * CIN + c) << 12) + (hy << 6) + wx];
            sA[m][i] = f2bf(v);
        }
        // Stage B: pre-converted bf16 weights, 16B chunks (128 chunks total)
        {
            const int n  = tid >> 2;
            const int i8 = (tid & 3) * 8;
            *(uint4*)&sB[n][i8] = *(const uint4*)&wom_bf[n * KDIM + kk0 + i8];
        }
        __syncthreads();

        FragBF fa;
        fa.q[0] = *(const uint4*)&sA[wv * 16 + mrow][kbA];
        fa.q[1] = *(const uint4*)&sA[wv * 16 + mrow][kbA + 8];
        FragBF fb0, fb1;
        fb0.q[0] = *(const uint4*)&sB[mrow][kbB];
        fb0.q[1] = *(const uint4*)&sB[mrow][kbB + 8];
        fb1.q[0] = *(const uint4*)&sB[16 + mrow][kbB];
        fb1.q[1] = *(const uint4*)&sB[16 + mrow][kbB + 8];

        acc0 = __builtin_amdgcn_wmma_f32_16x16x32_bf16(
            false, fa.v, false, fb0.v, (short)0, acc0, false, false);
        acc1 = __builtin_amdgcn_wmma_f32_16x16x32_bf16(
            false, fa.v, false, fb1.v, (short)0, acc1, false, false);
        __syncthreads();
    }

    const int mhi = (lane < 16) ? 0 : 8;
    for (int r = 0; r < 8; ++r) {
        const int pix = pix_base + wv * 16 + r + mhi;
        int ch = mrow;
        if (ch < OMCH) om[((b * OMCH + ch) << 12) + pix] = acc0[r] + b_om[ch];
        ch = 16 + mrow;
        if (ch < OMCH) om[((b * OMCH + ch) << 12) + pix] = acc1[r] + b_om[ch];
    }
}

// ---------------------------------------------------------------------------
// Kernel 2: fused deformable sampling + main contraction as WMMA GEMM.
// Workgroup: 64 pixels x ALL 256 output channels. Bilinear table (mask &
// OOB folded) precomputed per (pixel, tap). B tile staged into LDS with
// GLOBAL_LOAD_ASYNC_TO_LDS_B128 (ASYNCcnt), A tile computed via 4-gather
// bilinear sampling. 512 threads = 16 waves, 4 WMMA tiles per wave.
// Grid: (64 pixel-groups, 8 batches).
// ---------------------------------------------------------------------------
__global__ __launch_bounds__(512) void dcn_deform_gemm_wmma(
    const float* __restrict__ x, const float* __restrict__ om,
    const unsigned short* __restrict__ wdc_bf, const float* __restrict__ b_dc,
    float* __restrict__ out)
{
    __shared__ __align__(16) unsigned short sA[64][32];    // pixels x kk (bf16)
    __shared__ __align__(16) unsigned short sB[256][32];   // n x kk (bf16)
    __shared__ int   sIdx[64 * KTAP][4];                   // clamped gather idx
    __shared__ float sWgt[64 * KTAP][4];                   // bilinear*valid*mask

    const int b = blockIdx.y;
    const int pix_base = blockIdx.x * 64;
    const int tid  = threadIdx.x;
    const int wv   = tid >> 5;
    const int lane = tid & 31;

    // --- Precompute per-(pixel, tap) sampling table ---
    for (int e = tid; e < 64 * KTAP; e += 512) {
        const int m = e / 9, k = e - 9 * m;
        const int p = pix_base + m;
        const int hy = p >> 6, wx = p & 63;
        const float dy = om[((b * OMCH + 2 * k) << 12) + p];
        const float dx = om[((b * OMCH + 2 * k + 1) << 12) + p];
        const float ml = om[((b * OMCH + 18 + k) << 12) + p];
        const float mask = 1.0f / (1.0f + __expf(-ml));
        const float py = (float)(hy - 1 + (k / 3)) + dy;
        const float px = (float)(wx - 1 + (k - 3 * (k / 3))) + dx;
        const float y0f = floorf(py), x0f = floorf(px);
        const int y0 = (int)y0f, x0 = (int)x0f;
        const float wy1 = py - y0f, wx1 = px - x0f;
        const float wys[2] = { 1.f - wy1, wy1 };
        const float wxs[2] = { 1.f - wx1, wx1 };
        const int   ys[2]  = { y0, y0 + 1 };
        const int   xs[2]  = { x0, x0 + 1 };
        #pragma unroll
        for (int j = 0; j < 4; ++j) {
            const int yy = ys[j >> 1], xx = xs[j & 1];
            const bool valid = (yy >= 0) && (yy < 64) && (xx >= 0) && (xx < 64);
            const int yc = min(max(yy, 0), 63);
            const int xc = min(max(xx, 0), 63);
            sIdx[e][j] = (yc << 6) + xc;
            sWgt[e][j] = valid ? (wys[j >> 1] * wxs[j & 1] * mask) : 0.f;
        }
    }
    __syncthreads();

    v8f acc[4] = {};
    const int mt   = wv & 3;          // which 16-pixel tile (0..3)
    const int ntg  = (wv >> 2) * 4;   // first of 4 owned n-tiles
    const int mrow = lane & 15;
    const int kbA  = (lane < 16) ? 0 : 16;
    const int kbB  = (lane < 16) ? 0 : 16;

    for (int it = 0; it < KSTEPS; ++it) {
        const int kk0 = it * 32;

        // Stage B: async global->LDS, 1024 x 16B chunks, 2 per thread.
        #pragma unroll
        for (int e = tid; e < 1024; e += 512) {
            const int n  = e >> 2;
            const int i8 = (e & 3) * 8;
            const unsigned int lds_addr =
                (unsigned int)(unsigned long long)(uintptr_t)&sB[n][i8];
            const unsigned long long gaddr =
                (unsigned long long)(uintptr_t)&wdc_bf[n * KDIM + kk0 + i8];
            asm volatile("global_load_async_to_lds_b128 %0, %1, off"
                         :: "v"(lds_addr), "v"(gaddr) : "memory");
        }

        // Stage A: deformable-sampled activations, 64 pixels x 32 kk
        for (int e = tid; e < 2048; e += 512) {
            const int m = e >> 5, i = e & 31;
            const int kk = kk0 + i;
            const int c = kk / 9, k = kk - 9 * c;
            const int t = m * 9 + k;
            const float* xb = x + ((b * CIN + c) << 12);
            float v = sWgt[t][0] * xb[sIdx[t][0]]
                    + sWgt[t][1] * xb[sIdx[t][1]]
                    + sWgt[t][2] * xb[sIdx[t][2]]
                    + sWgt[t][3] * xb[sIdx[t][3]];
            sA[m][i] = f2bf(v);
        }

        asm volatile("s_wait_asynccnt 0x0" ::: "memory");
        __syncthreads();

        FragBF fa;
        fa.q[0] = *(const uint4*)&sA[mt * 16 + mrow][kbA];
        fa.q[1] = *(const uint4*)&sA[mt * 16 + mrow][kbA + 8];
        #pragma unroll
        for (int j = 0; j < 4; ++j) {
            FragBF fb;
            const int n = (ntg + j) * 16 + mrow;
            fb.q[0] = *(const uint4*)&sB[n][kbB];
            fb.q[1] = *(const uint4*)&sB[n][kbB + 8];
            acc[j] = __builtin_amdgcn_wmma_f32_16x16x32_bf16(
                false, fa.v, false, fb.v, (short)0, acc[j], false, false);
        }
        __syncthreads();
    }

    const int mhi = (lane < 16) ? 0 : 8;
    #pragma unroll
    for (int j = 0; j < 4; ++j) {
        const int ch = (ntg + j) * 16 + mrow;
        const float bias = b_dc[ch];
        #pragma unroll
        for (int r = 0; r < 8; ++r) {
            const int pix = pix_base + mt * 16 + r + mhi;
            out[((b * COUT + ch) << 12) + pix] = acc[j][r] + bias;
        }
    }
}

// ---------------------------------------------------------------------------
extern "C" void kernel_launch(void* const* d_in, const int* in_sizes, int n_in,
                              void* d_out, int out_size, void* d_ws, size_t ws_size,
                              hipStream_t stream) {
    (void)in_sizes; (void)n_in; (void)out_size; (void)ws_size;
    const float* x    = (const float*)d_in[0];
    const float* w_om = (const float*)d_in[1];
    const float* b_om = (const float*)d_in[2];
    const float* w_dc = (const float*)d_in[3];
    const float* b_dc = (const float*)d_in[4];
    float* out = (float*)d_out;

    char* ws = (char*)d_ws;
    float* om = (float*)(ws + OM_OFF);                          // 3.46 MB
    unsigned short* wom_bf = (unsigned short*)(ws + WOMBF_OFF); // 144 KB
    unsigned short* wdc_bf = (unsigned short*)(ws + WDCBF_OFF); // 1.13 MB

    const int conv_elems = WOM_ELEMS + WDC_ELEMS;
    dcn_convert_weights<<<(conv_elems + 255) / 256, 256, 0, stream>>>(
        w_om, w_dc, wom_bf, wdc_bf);
    dcn_offset_conv_wmma<<<dim3(64, 8), 128, 0, stream>>>(x, wom_bf, b_om, om);
    dcn_deform_gemm_wmma<<<dim3(64, 8), 512, 0, stream>>>(x, om, wdc_bf, b_dc, out);
}
